// Beran_54150947668145
// MI455X (gfx1250) — matrix operand order repaired and checked by reading, not compiled, not run on hardware
//
#include <hip/hip_runtime.h>
#include <hip/hip_bf16.h>
#include <math.h>

typedef __attribute__((ext_vector_type(16))) _Float16 v16h;
typedef __attribute__((ext_vector_type(8)))  float    v8f;

#define Kc 16
#define Bb 512
#define Nn 4096
#define Vv 32
#define KV 512   /* Kc*Vv : GEMM inner dimension */

/* float-element offsets of scratch staged inside d_out's first half
   (overwritten by surv_func at the end):
   Pf16  : 512*512 halfs = 131072 floats at offset 0
   sumsq : 16*512 floats  at 131072
   base  : 512 floats     at 139264                                   */
#define SUMSQ_OFF 131072
#define BASE_OFF  139264

/* ---------------- Kernel 1: softmax over V=32, one wave per (k,b) row -------- */
__global__ void k_softmax_prep(const float* __restrict__ cp, float* __restrict__ outbuf) {
    const int tid  = threadIdx.x;
    const int lane = tid & 31;
    const int wv   = tid >> 5;
    const int row  = blockIdx.x * 8 + wv;        /* row in [0, K*B) */
    const int k    = row >> 9;                   /* / 512 */
    const int b    = row & 511;

    float l = cp[row * Vv + lane];
    float m = l;
    #pragma unroll
    for (int off = 16; off > 0; off >>= 1) m = fmaxf(m, __shfl_xor(m, off, 32));
    float e = __expf(l - m);
    float s = e;
    #pragma unroll
    for (int off = 16; off > 0; off >>= 1) s += __shfl_xor(s, off, 32);
    float p = e / s;

    _Float16* pf = reinterpret_cast<_Float16*>(outbuf);
    pf[b * KV + k * Vv + lane] = (_Float16)p;    /* Pf16 row-major [B, 512] */

    float sq = p * p;
    #pragma unroll
    for (int off = 16; off > 0; off >>= 1) sq += __shfl_xor(sq, off, 32);
    if (lane == 0) outbuf[SUMSQ_OFF + k * Bb + b] = sq;
}

/* ---------------- Kernel 2: base[b] = K + sum_k sumsq[k][b] ------------------ */
__global__ void k_base_reduce(float* __restrict__ outbuf) {
    int b = blockIdx.x * blockDim.x + threadIdx.x;
    if (b < Bb) {
        float acc = (float)Kc;
        #pragma unroll
        for (int k = 0; k < Kc; ++k) acc += outbuf[SUMSQ_OFF + k * Bb + b];
        outbuf[BASE_OFF + b] = acc;
    }
}

/* ---------------- Kernel 3: metric GEMM via WMMA + exp ----------------------- */
/* D[b,n] = sum_kv P[b,kv] * onehot(c_in[n,k]==v); one 16x16 tile per wave.
   A (f16 16x32) fragments read from an LDS panel shared by 8 waves; the one-hot
   B fragments are synthesized in registers from c_in labels.                   */
__global__ void k_metric_wmma(const int* __restrict__ cin,
                              const float* __restrict__ bwp,
                              float* __restrict__ outbuf) {
    __shared__ __align__(16) _Float16 panel[16 * KV];   /* 16 KB A-panel */

    const int tid = threadIdx.x;
    const int m0  = blockIdx.y * 16;                    /* b block row */

    /* cooperative load: rows m0..m0+15 of Pf16 are 8192 contiguous halfs */
    {
        const uint4* src = reinterpret_cast<const uint4*>(
            reinterpret_cast<const _Float16*>(outbuf) + (size_t)m0 * KV);
        uint4* dst = reinterpret_cast<uint4*>(panel);
        #pragma unroll
        for (int i = 0; i < 4; ++i) dst[tid + 256 * i] = src[tid + 256 * i];
    }
    __syncthreads();

    const int lane = tid & 31;
    const int wv   = tid >> 5;
    const int hl   = lane >> 4;      /* lane-half: 0 or 1 */
    const int Ml   = lane & 15;      /* M (A) / N (B,C,D) index within tile */
    const int n0   = (blockIdx.x * 8 + wv) * 16;

    /* per-lane labels for its n = n0 + Ml (duplicated across lane halves) */
    int labs[16];
    {
        const int4* crow = reinterpret_cast<const int4*>(cin + (size_t)(n0 + Ml) * Kc);
        #pragma unroll
        for (int i = 0; i < 4; ++i) {
            int4 t = crow[i];
            labs[4*i+0] = t.x; labs[4*i+1] = t.y; labs[4*i+2] = t.z; labs[4*i+3] = t.w;
        }
    }

    v8f c = {};
    const _Float16* prow = &panel[Ml * KV];

    #pragma unroll
    for (int kc = 0; kc < 16; ++kc) {        /* 16 chunks of 32 over inner K=512 */
        const int kk = kc * 32;
        v16h a, bf;
        /* ISA 16-bit A 16x32 layout: VGPRs 0-3 hold K = 8*half + 0..7,
           VGPRs 4-7 hold K = 16 + 8*half + 0..7 (two contiguous 8-half runs) */
        #pragma unroll
        for (int j = 0; j < 8; ++j) {
            a[j]     = prow[kk + 8*hl + j];
            a[8 + j] = prow[kk + 16 + 8*hl + j];
        }
        /* B 32x16: same K order as A (j<8: K=8*hl+j, j>=8: K=16+8*hl+j-8); since kk is a
           multiple of 32, the concept index k is uniform (= kc) for the fragment,
           and v = that K  ->  one-hot compare against labs[kc].               */
        const int lab = labs[kc];
        #pragma unroll
        for (int j = 0; j < 16; ++j)
            bf[j] = (lab == (j < 8 ? hl * 8 + j : 8 + hl * 8 + j)) ? (_Float16)1.0f : (_Float16)0.0f;

        c = __builtin_amdgcn_wmma_f32_16x16x32_f16(
                /*neg_a=*/false, a, /*neg_b=*/false, bf,
                /*c_mod=*/(short)0, c, /*reuse_a=*/false, /*reuse_b=*/false);
    }

    const float bw    = fminf(fmaxf(bwp[0], 0.1f), 10.0f);
    const float invbw = 1.0f / bw;
    float* wout = outbuf + (size_t)Bb * Nn;          /* w lives in steps half */

    /* f32 C/D layout: VGPR r -> M = r + 8*half ; N = lane & 15 */
    #pragma unroll
    for (int r = 0; r < 8; ++r) {
        const int b     = m0 + r + 8 * hl;
        const float bse = outbuf[BASE_OFF + b];
        const float metric = bse - 2.0f * c[r];
        wout[(size_t)b * Nn + n0 + Ml] = __expf(-metric * invbw);
    }
}

/* ---------------- Kernel 4: Beran estimator scan, one WG per b row ----------- */
__device__ __forceinline__ float block_iscan(float x, float* lds, int tid, float& total) {
    const int lane = tid & 31, wv = tid >> 5;
    #pragma unroll
    for (int off = 1; off < 32; off <<= 1) {
        float v = __shfl_up(x, off, 32);
        if (lane >= off) x += v;
    }
    __syncthreads();                 /* protect lds reuse from previous call */
    if (lane == 31) lds[wv] = x;
    __syncthreads();
    float pre = 0.f, tot = 0.f;
    #pragma unroll
    for (int i = 0; i < 8; ++i) {
        float wi = lds[i];
        tot += wi;
        if (i < wv) pre += wi;
    }
    total = tot;
    return x + pre;
}

__global__ void k_beran_scan(const float* __restrict__ delta, float* __restrict__ outbuf) {
    __shared__ float lds[8];
    __shared__ float bc;
    const int tid = threadIdx.x, lane = tid & 31, wv = tid >> 5;
    const int b = blockIdx.x;
    float* wrow = outbuf + (size_t)Bb * Nn + (size_t)b * Nn;  /* w -> steps (in place) */
    float* srow = outbuf + (size_t)b * Nn;                    /* surv_func            */

    /* pass 1: s = sum_n w */
    float acc = 0.f;
    for (int t = 0; t < 16; ++t) acc += wrow[t * 256 + tid];
    #pragma unroll
    for (int off = 16; off > 0; off >>= 1) acc += __shfl_xor(acc, off, 32);
    if (lane == 0) lds[wv] = acc;
    __syncthreads();
    if (tid == 0) { float s = 0.f; for (int i = 0; i < 8; ++i) s += lds[i]; bc = s; }
    __syncthreads();
    const float s = bc;
    const float wscale = (s < 1e-13f) ? 0.f : (1.f / s);

    const float tol = 1e-8f + 1e-5f;   /* jnp.isclose(x, 1.0) tolerance */
    float carryW = 0.f, carryH = 0.f, s2acc = 0.f;

    for (int t = 0; t < 16; ++t) {
        const int n = t * 256 + tid;
        const float Wv = wrow[n] * wscale;
        float totW, totY;
        const float sc  = block_iscan(Wv, lds, tid, totW);
        const float wc  = carryW + sc;
        const float sh  = wc - Wv;
        const bool bad  = (fabsf(sh - 1.f) <= tol) || (fabsf(wc - 1.f) <= tol);
        const float sh2 = bad ? 0.f : sh;
        const float wc2 = bad ? 0.f : wc;
        const float xi  = logf(1.f - sh2) - logf(1.f - wc2);
        const float y   = delta[n] * xi;
        const float sy  = block_iscan(y, lds, tid, totY);
        const float haz = carryH + sy;
        const float surv = __expf(-haz);
        /* steps[n] = surv[n-1]-surv[n], with surv[-1]=1: exp(-(haz-y)) is exactly
           surv[n-1], and equals 1 at n==0 -> matches the first_step concat. */
        const float step = __expf(-(haz - y)) - surv;
        srow[n] = surv;
        wrow[n] = step;
        s2acc  += step;
        carryW += totW;
        carryH += totY;
    }

    __syncthreads();
    #pragma unroll
    for (int off = 16; off > 0; off >>= 1) s2acc += __shfl_xor(s2acc, off, 32);
    if (lane == 0) lds[wv] = s2acc;
    __syncthreads();
    if (tid == 0) { float s2 = 0.f; for (int i = 0; i < 8; ++i) s2 += lds[i]; bc = s2; }
    __syncthreads();
    const float s2 = bc;
    const float scale2 = (s2 < 1e-13f) ? 0.f : (1.f / s2);
    for (int t = 0; t < 16; ++t) wrow[t * 256 + tid] *= scale2;
}

extern "C" void kernel_launch(void* const* d_in, const int* in_sizes, int n_in,
                              void* d_out, int out_size, void* d_ws, size_t ws_size,
                              hipStream_t stream) {
    (void)in_sizes; (void)n_in; (void)out_size; (void)d_ws; (void)ws_size;
    const int*   c_in  = (const int*)  d_in[0];   /* [N, K] int32  */
    const float* delta = (const float*)d_in[1];   /* [N]           */
    const float* c_p   = (const float*)d_in[2];   /* [K, B, V]     */
    const float* bw    = (const float*)d_in[3];   /* [1]           */
    float* out = (float*)d_out;                   /* surv_func | surv_steps */

    k_softmax_prep<<<dim3((Kc * Bb) / 8), dim3(256), 0, stream>>>(c_p, out);
    k_base_reduce <<<dim3(2),             dim3(256), 0, stream>>>(out);
    k_metric_wmma <<<dim3(32, 32),        dim3(256), 0, stream>>>(c_in, bw, out);
    k_beran_scan  <<<dim3(Bb),            dim3(256), 0, stream>>>(delta, out);
}
